// DualScaleOceanAttention_9268539425471
// MI455X (gfx1250) — matrix operand order repaired and checked by reading, not compiled, not run on hardware
//
#include <hip/hip_runtime.h>

// ============================================================================
// MaxViT dual (window + grid) attention block for MI455X (gfx1250).
//
// All matmuls on v_wmma_f32_16x16x32_bf16 with a bf16 hi/lo split ("bf16x3")
// for near-fp32 accuracy on the large-K reductions.  GEMMs stream from global
// (L2 = 192MB holds everything hot); attention uses per-wave LDS scratch for
// scores / softmax / P(bf16).  Round-2 changes: native __bf16 conversions
// (hardware cvt ops instead of manual RNE bit-twiddling -> much less VALU),
// and global_prefetch on the streaming GEMM operand.
// ============================================================================

typedef __bf16  v16bf __attribute__((ext_vector_type(16)));
typedef __bf16  v8bf  __attribute__((ext_vector_type(8)));
typedef float   v8f   __attribute__((ext_vector_type(8)));

#define DEVINL static __device__ __forceinline__

// ---------------- bf16 helpers (native hardware converts) -------------------
DEVINL __bf16 f2bf(float f) { return (__bf16)f; }

DEVINL v8f zero8() {
  v8f z = {0.f, 0.f, 0.f, 0.f, 0.f, 0.f, 0.f, 0.f};
  return z;
}

// ---------------- WMMA wrappers --------------------------------------------
DEVINL v8f wmma_bf16(v16bf a, v16bf b, v8f c) {
  return __builtin_amdgcn_wmma_f32_16x16x32_bf16(false, a, false, b,
                                                 (short)0, c, false, false);
}
// hi/lo split product: (ah+al)*(bh+bl) ~= ah*bh + ah*bl + al*bh
DEVINL v8f mma_x3(v16bf ah, v16bf al, v16bf bh, v16bf bl, v8f c) {
  c = wmma_bf16(ah, bh, c);
  c = wmma_bf16(ah, bl, c);
  c = wmma_bf16(al, bh, c);
  return c;
}

// ---------------- fragment loaders -----------------------------------------
// A fragment (16x32 bf16): lane r<16 -> row r, K = half*16 .. half*16+15
DEVINL void load_a_f32(const float* __restrict__ p0, int ld, int lane,
                       v16bf& hi, v16bf& lo) {
  const float* p = p0 + (lane & 15) * ld + ((lane >> 4) & 1) * 16;
#pragma unroll
  for (int j = 0; j < 8; ++j) {
    float f = p[j];
    __bf16 h = (__bf16)f;
    hi[j] = h;
    lo[j] = (__bf16)(f - (float)h);
    float g = p[8 + j];
    __bf16 h2 = (__bf16)g;
    hi[8 + j] = h2;
    lo[8 + j] = (__bf16)(g - (float)h2);
  }
}

// B fragment (32x16 bf16): lane r<16 -> col r, K = half*16 .. half*16+15
DEVINL void load_b_f32(const float* __restrict__ p0, int ld, int lane,
                       v16bf& hi, v16bf& lo) {
  const float* p = p0 + (lane & 15) * ld + ((lane >> 4) & 1) * 16;
#pragma unroll
  for (int j = 0; j < 16; ++j) {
    float f = p[j];
    __bf16 h = (__bf16)f;
    hi[j] = h;
    lo[j] = (__bf16)(f - (float)h);
  }
}

// B fragment, bf16 source (used for V, stored transposed [d][t])
DEVINL v16bf load_b_bf16(const __bf16* __restrict__ p0, int ld, int lane) {
  const __bf16* p = p0 + (lane & 15) * ld + ((lane >> 4) & 1) * 16;
  v8bf a = *(const v8bf*)p;
  v8bf b = *(const v8bf*)(p + 8);
  return __builtin_shufflevector(a, b, 0, 1, 2, 3, 4, 5, 6, 7,
                                 8, 9, 10, 11, 12, 13, 14, 15);
}

// A fragment, bf16 LDS source (used for P), rows pitch in elements
DEVINL v16bf load_a_bf16_lds(const __bf16* base, int pitch, int lane, int kc) {
  const __bf16* p = base + (lane & 15) * pitch + kc * 32 + ((lane >> 4) & 1) * 16;
  v8bf a = *(const v8bf*)p;
  v8bf b = *(const v8bf*)(p + 8);
  return __builtin_shufflevector(a, b, 0, 1, 2, 3, 4, 5, 6, 7,
                                 8, 9, 10, 11, 12, 13, 14, 15);
}

// ============================================================================
// Problem constants: B=8, H=W=128, C=256, 8 heads * hd 32, window 8 (no pad)
// local pairs  : (b,h,win)  = 8*4*256 = 8192, 64 tokens each
// global pairs : (b,hg,grid)= 8*4*64  = 2048, 256 tokens each
// ============================================================================

// ---------------- kernel 0: transpose weights so B-frag loads are contiguous
__global__ void wtrans_kernel(const float* __restrict__ src,
                              float* __restrict__ dst, int K, int N) {
  int i = blockIdx.x * blockDim.x + threadIdx.x;
  if (i < K * N) {
    int k = i / N, n = i % N;
    dst[n * K + k] = src[i];
  }
}

// ---------------- kernel 1: QKV GEMM + scatter into attention layouts ------
// x[131072,256] * W[256,768]; wave tile = 16 rows x 64 cols, K loop 8x32.
__global__ void qkv_gemm_kernel(const float* __restrict__ x,
                                const float* __restrict__ wT,
                                float* __restrict__ Qw, float* __restrict__ Kw,
                                __bf16* __restrict__ Vw,
                                float* __restrict__ Qg, float* __restrict__ Kg,
                                __bf16* __restrict__ Vg) {
  int wave = threadIdx.x >> 5, lane = threadIdx.x & 31;
  int gid = blockIdx.x * 8 + wave;          // 98304 wave units
  int mtile = gid / 12, ng = gid % 12;
  int row0 = mtile * 16, nb = ng * 64;

  v8f acc[4];
#pragma unroll
  for (int nt = 0; nt < 4; ++nt) acc[nt] = zero8();

  for (int kc = 0; kc < 8; ++kc) {
    // prefetch next K-chunk of the streaming operand (global_prefetch_b8)
    if (kc < 7)
      __builtin_prefetch(x + row0 * 256 + (kc + 1) * 32 + (lane & 15) * 256, 0, 1);
    v16bf ah, al;
    load_a_f32(x + row0 * 256 + kc * 32, 256, lane, ah, al);
#pragma unroll
    for (int nt = 0; nt < 4; ++nt) {
      v16bf bh, bl;
      load_b_f32(wT + (nb + nt * 16) * 256 + kc * 32, 256, lane, bh, bl);
      acc[nt] = mma_x3(ah, al, bh, bl, acc[nt]);
    }
  }

  int half = (lane >> 4) & 1, ln = lane & 15;
#pragma unroll
  for (int j = 0; j < 8; ++j) {
    int tok = row0 + half * 8 + j;
    int b = tok >> 14, n = tok & 16383, y = n >> 7, xx = n & 127;
#pragma unroll
    for (int nt = 0; nt < 4; ++nt) {
      int c = nb + nt * 16 + ln;
      int sel = c >> 8, cc = c & 255, h = cc >> 5, d = cc & 31;
      float v = acc[nt][j];
      if (h < 4) {  // window (local) heads
        int win = (y >> 3) * 16 + (xx >> 3);
        int t = (y & 7) * 8 + (xx & 7);
        int p = ((b * 4 + h) << 8) + win;          // pair id
        if (sel == 0)      Qw[p * 2048 + t * 32 + d] = v;
        else if (sel == 1) Kw[p * 2048 + t * 32 + d] = v;
        else               Vw[p * 2048 + d * 64 + t] = f2bf(v);  // transposed
      } else {      // grid (global) heads
        int g = (y & 7) * 8 + (xx & 7);
        int u = (y >> 3) * 16 + (xx >> 3);
        int p = ((b * 4 + (h - 4)) << 6) + g;
        if (sel == 0)      Qg[p * 8192 + u * 32 + d] = v;
        else if (sel == 1) Kg[p * 8192 + u * 32 + d] = v;
        else               Vg[p * 8192 + d * 256 + u] = f2bf(v);  // transposed
      }
    }
  }
}

// ---------------- kernel 2: local window attention -------------------------
// 1 wave = 1 (b,h,win) pair, 64x64 scores in LDS, softmax, PV, scatter out.
// per-wave LDS: 4096 f32 scores + 64x64 bf16 P (2048 f-equiv) + 64 sums.
__global__ void attn_local_kernel(const float* __restrict__ Qw,
                                  const float* __restrict__ Kw,
                                  const __bf16* __restrict__ Vw,
                                  float* __restrict__ attn) {
  extern __shared__ float lds[];
  int wave = threadIdx.x >> 5, lane = threadIdx.x & 31;
  int pair = blockIdx.x * 8 + wave;
  int b = pair >> 10, rem = pair & 1023, h = rem >> 8, win = rem & 255;
  int wy = win >> 4, wx = win & 15;
  float* S = lds + wave * 6208;
  __bf16* P = (__bf16*)(S + 4096);
  float* sums = S + 6144;
  const float scale = 0.17677669529663687f;  // 1/sqrt(32)
  const float* qb = Qw + pair * 2048;
  const float* kb = Kw + pair * 2048;
  const __bf16* vb = Vw + pair * 2048;
  int half = (lane >> 4) & 1, ln = lane & 15;

  // Q fragments resident (hd=32 == one K chunk)
  v16bf qh[4], ql[4];
#pragma unroll
  for (int mt = 0; mt < 4; ++mt)
    load_a_f32(qb + mt * 16 * 32, 32, lane, qh[mt], ql[mt]);

  // scores = Q K^T * scale -> LDS (bf16x3)
#pragma unroll
  for (int nt = 0; nt < 4; ++nt) {
    v16bf kh, kl;
    load_b_f32(kb + nt * 16 * 32, 32, lane, kh, kl);
#pragma unroll
    for (int mt = 0; mt < 4; ++mt) {
      v8f acc = mma_x3(qh[mt], ql[mt], kh, kl, zero8());
      int rb = mt * 16 + half * 8, col = nt * 16 + ln;
#pragma unroll
      for (int j = 0; j < 8; ++j) S[(rb + j) * 64 + col] = acc[j] * scale;
    }
  }
  __syncthreads();

  // softmax rows (2 rows / lane); store unnormalized exp as bf16 P
#pragma unroll
  for (int rr = 0; rr < 2; ++rr) {
    int row = lane + rr * 32;
    const float* sr = S + row * 64;
    float m = sr[0];
    for (int i = 1; i < 64; ++i) m = fmaxf(m, sr[i]);
    float sum = 0.f;
    __bf16* pr = P + row * 64;
    for (int i = 0; i < 64; ++i) {
      float e = __expf(sr[i] - m);
      sum += e;
      pr[i] = f2bf(e);
    }
    sums[row] = sum;
  }
  __syncthreads();

  // out = (P V) / rowsum, scatter to attn[token, h*32+d]
  for (int mt = 0; mt < 4; ++mt) {
#pragma unroll
    for (int nt = 0; nt < 2; ++nt) {
      v8f acc = zero8();
#pragma unroll
      for (int kc = 0; kc < 2; ++kc) {
        v16bf pa = load_a_bf16_lds(P + mt * 16 * 64, 64, lane, kc);
        v16bf vv = load_b_bf16(vb + nt * 16 * 64 + kc * 32, 64, lane);
        acc = wmma_bf16(pa, vv, acc);
      }
      int d = nt * 16 + ln;
#pragma unroll
      for (int j = 0; j < 8; ++j) {
        int row = mt * 16 + half * 8 + j;  // token in window
        float o = acc[j] / sums[row];
        int y = wy * 8 + (row >> 3), xx = wx * 8 + (row & 7);
        attn[(((b << 14) + y * 128 + xx) << 8) + h * 32 + d] = o;
      }
    }
  }
}

// ---------------- kernel 3: global grid attention --------------------------
// 1 block = 1 (b,hg,grid) pair of 256 tokens; wave = 32-row slab, processed
// as two 16-row halves to fit LDS (per-wave: 16x256 f32 + 16x256 bf16 + 16).
__global__ void attn_global_kernel(const float* __restrict__ Qg,
                                   const float* __restrict__ Kg,
                                   const __bf16* __restrict__ Vg,
                                   float* __restrict__ attn) {
  extern __shared__ float lds[];
  int wave = threadIdx.x >> 5, lane = threadIdx.x & 31;
  int pair = blockIdx.x;  // 2048
  int b = pair >> 8, rem = pair & 255, hg = rem >> 6, g = rem & 63;
  int iy = g >> 3, ix = g & 7;
  float* S = lds + wave * 6160;
  __bf16* P = (__bf16*)(S + 4096);
  float* sums = S + 6144;
  const float scale = 0.17677669529663687f;
  const float* qb = Qg + pair * 8192;
  const float* kb = Kg + pair * 8192;
  const __bf16* vb = Vg + pair * 8192;
  int half = (lane >> 4) & 1, ln = lane & 15;

  for (int mh = 0; mh < 2; ++mh) {
    int m0 = wave * 32 + mh * 16;
    v16bf ah, al;
    load_a_f32(qb + m0 * 32, 32, lane, ah, al);

    for (int nt = 0; nt < 16; ++nt) {
      v16bf kh, kl;
      load_b_f32(kb + nt * 16 * 32, 32, lane, kh, kl);
      v8f acc = mma_x3(ah, al, kh, kl, zero8());
      int rb = half * 8, col = nt * 16 + ln;
#pragma unroll
      for (int j = 0; j < 8; ++j) S[(rb + j) * 256 + col] = acc[j] * scale;
    }
    __syncthreads();

    if (lane < 16) {  // one row per lane
      int row = lane;
      const float* sr = S + row * 256;
      float m = sr[0];
      for (int i = 1; i < 256; ++i) m = fmaxf(m, sr[i]);
      float sum = 0.f;
      __bf16* pr = P + row * 256;
      for (int i = 0; i < 256; ++i) {
        float e = __expf(sr[i] - m);
        sum += e;
        pr[i] = f2bf(e);
      }
      sums[row] = sum;
    }
    __syncthreads();

#pragma unroll
    for (int nt = 0; nt < 2; ++nt) {
      v8f acc = zero8();
#pragma unroll
      for (int kc = 0; kc < 8; ++kc) {
        v16bf pa = load_a_bf16_lds(P, 256, lane, kc);
        v16bf vv = load_b_bf16(vb + nt * 16 * 256 + kc * 32, 256, lane);
        acc = wmma_bf16(pa, vv, acc);
      }
      int d = nt * 16 + ln;
#pragma unroll
      for (int j = 0; j < 8; ++j) {
        int rloc = half * 8 + j;
        float o = acc[j] / sums[rloc];
        int u = m0 + rloc;  // token in grid
        int wy = u >> 4, wx = u & 15;
        int y = wy * 8 + iy, xx = wx * 8 + ix;
        attn[(((b << 14) + y * 128 + xx) << 8) + (hg + 4) * 32 + d] = o;
      }
    }
    __syncthreads();  // before next half reuses S/P
  }
}

// ---------------- kernel 4: output projection + bias -----------------------
__global__ void proj_gemm_kernel(const float* __restrict__ attn,
                                 const float* __restrict__ wT,
                                 const float* __restrict__ bias,
                                 float* __restrict__ out) {
  int wave = threadIdx.x >> 5, lane = threadIdx.x & 31;
  int gid = blockIdx.x * 8 + wave;  // 32768
  int mtile = gid >> 2, ng = gid & 3;
  int row0 = mtile * 16, nb = ng * 64;

  v8f acc[4];
#pragma unroll
  for (int nt = 0; nt < 4; ++nt) acc[nt] = zero8();

  for (int kc = 0; kc < 8; ++kc) {
    if (kc < 7)
      __builtin_prefetch(attn + row0 * 256 + (kc + 1) * 32 + (lane & 15) * 256, 0, 1);
    v16bf ah, al;
    load_a_f32(attn + row0 * 256 + kc * 32, 256, lane, ah, al);
#pragma unroll
    for (int nt = 0; nt < 4; ++nt) {
      v16bf bh, bl;
      load_b_f32(wT + (nb + nt * 16) * 256 + kc * 32, 256, lane, bh, bl);
      acc[nt] = mma_x3(ah, al, bh, bl, acc[nt]);
    }
  }

  int half = (lane >> 4) & 1, ln = lane & 15;
#pragma unroll
  for (int nt = 0; nt < 4; ++nt) {
    int c = nb + nt * 16 + ln;
    float bb = bias[c];
#pragma unroll
    for (int j = 0; j < 8; ++j) {
      int row = row0 + half * 8 + j;
      out[row * 256 + c] = acc[nt][j] + bb;
    }
  }
}

// ============================================================================
extern "C" void kernel_launch(void* const* d_in, const int* in_sizes, int n_in,
                              void* d_out, int out_size, void* d_ws,
                              size_t ws_size, hipStream_t stream) {
  (void)in_sizes; (void)n_in; (void)out_size; (void)ws_size;
  const float* x     = (const float*)d_in[0];   // [8,16384,256]
  const float* Wqkv  = (const float*)d_in[1];   // [256,768]
  const float* Wproj = (const float*)d_in[2];   // [256,256]
  const float* bproj = (const float*)d_in[3];   // [256]
  // d_in[4], d_in[5] are H=128, W=128 (hardcoded in the index math above)
  float* out = (float*)d_out;

  // workspace carve (~471 MB)
  char* p = (char*)d_ws;
  float* WqkvT  = (float*)p;  p += (size_t)768 * 256 * 4;
  float* WprojT = (float*)p;  p += (size_t)256 * 256 * 4;
  float*  Qw  = (float*)p;    p += (size_t)16777216 * 4;
  float*  Kw  = (float*)p;    p += (size_t)16777216 * 4;
  float*  Qg  = (float*)p;    p += (size_t)16777216 * 4;
  float*  Kg  = (float*)p;    p += (size_t)16777216 * 4;
  __bf16* Vw  = (__bf16*)p;   p += (size_t)16777216 * 2;
  __bf16* Vg  = (__bf16*)p;   p += (size_t)16777216 * 2;
  float* attn = (float*)p;    p += (size_t)33554432 * 4;

  // 0) transpose weights (tiny, L2-resident afterwards)
  wtrans_kernel<<<(768 * 256 + 255) / 256, 256, 0, stream>>>(Wqkv, WqkvT, 256, 768);
  wtrans_kernel<<<(256 * 256 + 255) / 256, 256, 0, stream>>>(Wproj, WprojT, 256, 256);

  // 1) QKV GEMM + scatter: 8192 M-tiles x 12 N-groups = 98304 waves
  qkv_gemm_kernel<<<12288, 256, 0, stream>>>(x, WqkvT, Qw, Kw, Vw, Qg, Kg, Vg);

  // 2) local window attention: 8192 pairs, 1 wave each
  size_t lds_local = (size_t)8 * 6208 * 4;   // ~198.7 KB / block (<= 320KB WGP)
  attn_local_kernel<<<1024, 256, lds_local, stream>>>(Qw, Kw, Vw, attn);

  // 3) global grid attention: 2048 pairs, 1 block each
  size_t lds_global = (size_t)8 * 6160 * 4;  // ~197.1 KB / block
  attn_global_kernel<<<2048, 256, lds_global, stream>>>(Qg, Kg, Vg, attn);

  // 4) projection: 8192 M-tiles x 4 N-groups = 32768 waves
  proj_gemm_kernel<<<4096, 256, 0, stream>>>(attn, WprojT, bproj, out);
}
